// PointMLP_10625749090928
// MI455X (gfx1250) — matrix operand order repaired and checked by reading, not compiled, not run on hardware
//
#include <hip/hip_runtime.h>

#define EPSF 1e-5f
#define KNN 32

typedef __attribute__((ext_vector_type(16))) __bf16 v16bf;
typedef __attribute__((ext_vector_type(8)))  __bf16 v8bf;
typedef __attribute__((ext_vector_type(8)))  float  v8f;

__device__ __forceinline__ __bf16 f2bf(float x) {
  unsigned u = __builtin_bit_cast(unsigned, x);
  unsigned r = u + 0x7FFFu + ((u >> 16) & 1u);   // round-to-nearest-even
  return __builtin_bit_cast(__bf16, (unsigned short)(r >> 16));
}

// ---------------------------------------------------------------------------
// GEMM: Y[M,N] = A[M,K] @ W[K,N] (+ bias), bf16 WMMA with f32 accumulate.
// Block tile 64x64, K-step 32, 4 waves, double-buffered LDS tiles.
// B tile stored K-transposed in LDS so fragments load as 2x ds_load_b128.
// ---------------------------------------------------------------------------
#define BM 64
#define BN 64
#define BK 32
#define SA_STRIDE 40   // elements; 80B rows -> 16B-aligned vector loads
#define SBT_STRIDE 40

__global__ __launch_bounds__(128)
void gemm_bf16_kernel(const float* __restrict__ A, const float* __restrict__ W,
                      const float* __restrict__ bias, float* __restrict__ Y,
                      int M, int K, int N) {
  __shared__ __attribute__((aligned(16))) __bf16 sA[2][BM * SA_STRIDE];
  __shared__ __attribute__((aligned(16))) __bf16 sB[2][BN * SBT_STRIDE]; // [n][k]
  const int tid  = threadIdx.x;
  const int wave = tid >> 5, lane = tid & 31;
  const int l15  = lane & 15, hs = lane >> 4;       // half-wave select
  const int m0 = blockIdx.y * BM, n0 = blockIdx.x * BN;

  v8f acc[4] = {};

  auto fill = [&](int k0, __bf16* dA, __bf16* dB) {
    // A tile: 64x32, row-major (K contiguous)
    for (int t = tid; t < BM * BK; t += 128) {
      int r = t >> 5, c = t & 31;
      int m = m0 + r, k = k0 + c;
      float v = (m < M && k < K) ? A[(size_t)m * K + k] : 0.f;
      dA[r * SA_STRIDE + c] = f2bf(v);
    }
    // B tile: stored transposed [n][k]; global read coalesced along n
    for (int t = tid; t < BK * BN; t += 128) {
      int n = t & 63, k = t >> 6;
      int gk = k0 + k, gn = n0 + n;
      float v = (gk < K && gn < N) ? W[(size_t)gk * N + gn] : 0.f;
      dB[n * SBT_STRIDE + k] = f2bf(v);
    }
  };

  fill(0, sA[0], sB[0]);
  __syncthreads();

  for (int k0 = 0; k0 < K; k0 += BK) {
    const int cur = (k0 / BK) & 1, nxt = cur ^ 1;

    // stage next K-tile while computing on the current one
    if (k0 + BK < K) fill(k0 + BK, sA[nxt], sB[nxt]);
    // warm GL2 for the tile after that (speculative prefetch)
    if (k0 + 2 * BK < K) {
      __builtin_prefetch(&A[(size_t)(m0 + wave * 16 + l15) * K + k0 + 2 * BK], 0, 1);
      __builtin_prefetch(&W[(size_t)(k0 + 2 * BK + wave * 8 + hs * 4) * N + n0 + l15 * 4], 0, 1);
    }

    // A fragment 16x32 (ISA 7.12.2): lanes0-15 K={0..7,16..23}, lanes16-31 K={8..15,24..31}
    const __bf16* pa = &sA[cur][(wave * 16 + l15) * SA_STRIDE + hs * 8];
    v8bf a0 = *(const v8bf*)pa;
    v8bf a1 = *(const v8bf*)(pa + 16);
    v16bf a = __builtin_shufflevector(a0, a1, 0, 1, 2, 3, 4, 5, 6, 7,
                                      8, 9, 10, 11, 12, 13, 14, 15);
    #pragma unroll
    for (int nt = 0; nt < 4; ++nt) {
      // B fragment 32x16: lanes0-15 K={0..7,16..23}, lanes16-31 K={8..15,24..31}, lane%16 = N
      const __bf16* pb = &sB[cur][(nt * 16 + l15) * SBT_STRIDE + hs * 8];
      v8bf b0 = *(const v8bf*)pb;
      v8bf b1 = *(const v8bf*)(pb + 16);
      v16bf b = __builtin_shufflevector(b0, b1, 0, 1, 2, 3, 4, 5, 6, 7,
                                        8, 9, 10, 11, 12, 13, 14, 15);
      acc[nt] = __builtin_amdgcn_wmma_f32_16x16x32_bf16(
          false, a, false, b, (short)0, acc[nt], false, false);
    }
    __syncthreads();
  }

  // C/D layout: VGPR r -> M = r (lanes 0-15) / r+8 (lanes 16-31), N = lane%16
  #pragma unroll
  for (int nt = 0; nt < 4; ++nt) {
    #pragma unroll
    for (int r = 0; r < 8; ++r) {
      int row = m0 + wave * 16 + r + hs * 8;
      int col = n0 + nt * 16 + l15;
      if (row < M && col < N)
        Y[(size_t)row * N + col] = acc[nt][r] + (bias ? bias[col] : 0.f);
    }
  }
}

// ---------------------------------------------------------------------------
// Per-column batch statistics (training-mode BN): one block per channel.
// ---------------------------------------------------------------------------
__global__ void colstats_kernel(const float* __restrict__ Y, float* __restrict__ mean,
                                float* __restrict__ var, int M, int N) {
  __shared__ float s1[256], s2[256];
  int col = blockIdx.x;
  float s = 0.f, ss = 0.f;
  for (int r = threadIdx.x; r < M; r += blockDim.x) {
    float v = Y[(size_t)r * N + col];
    s += v; ss += v * v;
  }
  s1[threadIdx.x] = s; s2[threadIdx.x] = ss;
  __syncthreads();
  for (int o = 128; o > 0; o >>= 1) {
    if ((int)threadIdx.x < o) {
      s1[threadIdx.x] += s1[threadIdx.x + o];
      s2[threadIdx.x] += s2[threadIdx.x + o];
    }
    __syncthreads();
  }
  if (threadIdx.x == 0) {
    float m = s1[0] / (float)M;
    mean[col] = m;
    var[col]  = fmaxf(s2[0] / (float)M - m * m, 0.f);
  }
}

// BN apply + optional residual + optional ReLU (elementwise, in-place safe)
__global__ void bn_act_kernel(const float* __restrict__ X, float* __restrict__ Y,
                              const float* __restrict__ mean, const float* __restrict__ var,
                              const float* __restrict__ g, const float* __restrict__ beta,
                              const float* __restrict__ res, int M, int N, int relu) {
  long long i = (long long)blockIdx.x * blockDim.x + threadIdx.x;
  long long tot = (long long)M * N;
  if (i >= tot) return;
  int c = (int)(i % N);
  float v = g[c] * (X[i] - mean[c]) * rsqrtf(var[c] + EPSF) + beta[c];
  if (res) v += res[i];
  if (relu) v = fmaxf(v, 0.f);
  Y[i] = v;
}

// out[b,s,:] = in[b, idx[b,s], :]
__global__ void gather_kernel(const float* __restrict__ in, const int* __restrict__ idx,
                              float* __restrict__ out, int B, int Nin, int S, int C) {
  int t = blockIdx.x * blockDim.x + threadIdx.x;
  int total = B * S * C;
  if (t >= total) return;
  int c = t % C, bs = t / C, b = bs / S;
  out[t] = in[((size_t)b * Nin + idx[bs]) * C + c];
}

// One block per query point: d2 table in LDS + 32 serial argmin passes
// (value tie -> lower index, matching lax.top_k).
__global__ void knn_kernel(const float* __restrict__ loc, const int* __restrict__ idx,
                           int* __restrict__ out, int B, int Nin, int S) {
  __shared__ float d2[1024];
  __shared__ float bv[256];
  __shared__ int   bi[256];
  __shared__ float q[3];
  int s = blockIdx.x, b = blockIdx.y;
  const float* Lb = loc + (size_t)b * Nin * 3;
  if (threadIdx.x == 0) {
    int qi = idx[b * S + s];
    q[0] = Lb[qi * 3 + 0]; q[1] = Lb[qi * 3 + 1]; q[2] = Lb[qi * 3 + 2];
  }
  __syncthreads();
  for (int n = threadIdx.x; n < Nin; n += blockDim.x) {
    float dx = Lb[n * 3 + 0] - q[0];
    float dy = Lb[n * 3 + 1] - q[1];
    float dz = Lb[n * 3 + 2] - q[2];
    d2[n] = dx * dx + dy * dy + dz * dz;
  }
  __syncthreads();
  for (int j = 0; j < KNN; ++j) {
    float best = __builtin_inff(); int besti = 0x7fffffff;
    for (int n = threadIdx.x; n < Nin; n += blockDim.x) {
      float v = d2[n];
      if (v < best) { best = v; besti = n; }   // ascending n => lowest index on tie
    }
    bv[threadIdx.x] = best; bi[threadIdx.x] = besti;
    __syncthreads();
    for (int o = 128; o > 0; o >>= 1) {
      if ((int)threadIdx.x < o) {
        float v2 = bv[threadIdx.x + o]; int i2 = bi[threadIdx.x + o];
        if (v2 < bv[threadIdx.x] || (v2 == bv[threadIdx.x] && i2 < bi[threadIdx.x])) {
          bv[threadIdx.x] = v2; bi[threadIdx.x] = i2;
        }
      }
      __syncthreads();
    }
    if (threadIdx.x == 0) {
      out[((size_t)b * S + s) * KNN + j] = bi[0];
      d2[bi[0]] = __builtin_inff();
    }
    __syncthreads();
  }
}

__global__ void zero_kernel(float* p, int n) {
  int t = blockIdx.x * blockDim.x + threadIdx.x;
  if (t < n) p[t] = 0.f;
}

// Gather neighborhood, subtract per-(b,s,c) mean over k, write centered values
// into X[:, 0:C+3], accumulate per-batch sum(d^2) for the ddof=1 std.
__global__ void group_center_kernel(const float* __restrict__ feats, const float* __restrict__ loc,
                                    const int* __restrict__ knn, float* __restrict__ X,
                                    float* __restrict__ ssq, int B, int Np, int C, int S, int XC) {
  int t = blockIdx.x * blockDim.x + threadIdx.x;
  int total = B * S * (C + 3);
  if (t >= total) return;
  int c = t % (C + 3);
  int bs = t / (C + 3);
  int b = bs / S;
  const int* kn = knn + (size_t)bs * KNN;
  float vals[KNN];
  if (c < C) {
    for (int j = 0; j < KNN; ++j) vals[j] = feats[((size_t)b * Np + kn[j]) * C + c];
  } else {
    int d = c - C;
    for (int j = 0; j < KNN; ++j) vals[j] = loc[((size_t)b * Np + kn[j]) * 3 + d];
  }
  float m = 0.f;
  for (int j = 0; j < KNN; ++j) m += vals[j];
  m *= (1.f / KNN);
  float acc = 0.f;
  float* Xp = X + (size_t)bs * KNN * XC + c;
  for (int j = 0; j < KNN; ++j) {
    float d0 = vals[j] - m;
    Xp[(size_t)j * XC] = d0;
    acc += d0 * d0;
  }
  atomicAdd(&ssq[b], acc);
}

__global__ void stdinv_kernel(const float* __restrict__ ssq, float* __restrict__ stdi,
                              float denom, int B) {
  int b = threadIdx.x;
  if (b < B) stdi[b] = 1.f / (sqrtf(ssq[b] / denom) + EPSF);
}

// alpha/beta affine on normalized part, anchor features into X[:, C+3 : 2C+3]
__global__ void finalize_x_kernel(float* __restrict__ X, const float* __restrict__ alpha,
                                  const float* __restrict__ beta, const float* __restrict__ stdi,
                                  const float* __restrict__ feats, const int* __restrict__ idx,
                                  int B, int S, int Np, int C, int XC) {
  long long t = (long long)blockIdx.x * blockDim.x + threadIdx.x;
  long long total = (long long)B * S * KNN * XC;
  if (t >= total) return;
  int c = (int)(t % XC);
  long long row = t / XC;
  int bs = (int)(row / KNN);
  int b = bs / S;
  if (c < C + 3) {
    X[t] = alpha[c] * (X[t] * stdi[b]) + beta[c];
  } else {
    int cc = c - (C + 3);
    X[t] = feats[((size_t)b * Np + idx[bs]) * C + cc];
  }
}

__global__ void maxpool_kernel(const float* __restrict__ Y, float* __restrict__ out,
                               int BS, int C) {
  int t = blockIdx.x * blockDim.x + threadIdx.x;
  if (t >= BS * C) return;
  int c = t % C, bs = t / C;
  const float* p = Y + (size_t)bs * KNN * C + c;
  float m = p[0];
  for (int j = 1; j < KNN; ++j) m = fmaxf(m, p[(size_t)j * C]);
  out[t] = m;
}

__global__ void gmax_kernel(const float* __restrict__ f, float* __restrict__ out,
                            int B, int S, int C) {
  int t = blockIdx.x * blockDim.x + threadIdx.x;
  if (t >= B * C) return;
  int c = t % C, b = t / C;
  const float* p = f + (size_t)b * S * C + c;
  float m = p[0];
  for (int s = 1; s < S; ++s) m = fmaxf(m, p[(size_t)s * C]);
  out[t] = m;
}

// ---------------------------------------------------------------------------
// Host orchestration
// ---------------------------------------------------------------------------
extern "C" void kernel_launch(void* const* d_in, const int* in_sizes, int n_in,
                              void* d_out, int out_size, void* d_ws, size_t ws_size,
                              hipStream_t stream) {
  (void)in_sizes; (void)n_in; (void)out_size; (void)ws_size;
  const int B = 8;
  const int counts[5] = {1024, 512, 256, 128, 64};
  const int chans[5]  = {64, 128, 256, 512, 1024};

  const float* points = (const float*)d_in[0];
  const int* idx0 = (const int*)d_in[1];
  const int* idxs[4] = {(const int*)d_in[2], (const int*)d_in[3],
                        (const int*)d_in[4], (const int*)d_in[5]};
  // params flattened in jax pytree (sorted-key) order:
  const float* bn1_beta = (const float*)d_in[6];
  const float* bn1_g    = (const float*)d_in[7];
  const float* bn2_beta = (const float*)d_in[8];
  const float* bn2_g    = (const float*)d_in[9];
  const float* fc1W = (const float*)d_in[10]; const float* fc1b = (const float*)d_in[11];
  const float* fc2W = (const float*)d_in[12]; const float* fc2b = (const float*)d_in[13];
  const float* fc3W = (const float*)d_in[14]; const float* fc3b = (const float*)d_in[15];
  const float* eW    = (const float*)d_in[16]; const float* eb = (const float*)d_in[17];
  const float* ebeta = (const float*)d_in[18]; const float* eg = (const float*)d_in[19];

  // workspace layout (floats)
  float* w = (float*)d_ws;
  size_t o = 0;
  float* locA  = w + o; o += 8 * 1024 * 3;
  float* locB  = w + o; o += 8 * 1024 * 3;
  float* featA = w + o; o += 524288;
  float* featB = w + o; o += 524288;
  float* poolA = w + o; o += 524288;
  float* poolB = w + o; o += 524288;
  int*   knn   = (int*)(w + o); o += 131072;
  float* meanb = w + o; o += 1024;
  float* varb  = w + o; o += 1024;
  float* ssq   = w + o; o += 16;
  float* stdi  = w + o; o += 16;
  float* X     = w + o; o += 17301504;   // also reused as GEMM scratch Y3
  float* Y1    = w + o; o += 16777216;
  float* Y2    = w + o; o += 16777216;

  auto gemm = [&](const float* Ap, const float* Wp, const float* bp, float* Yp,
                  int M, int K, int N) {
    dim3 g((N + BN - 1) / BN, (M + BM - 1) / BM);
    gemm_bf16_kernel<<<g, 128, 0, stream>>>(Ap, Wp, bp, Yp, M, K, N);
  };
  auto bn = [&](const float* Xp, float* Yp, const float* gp, const float* betap,
                const float* resp, int M, int N, int relu) {
    colstats_kernel<<<N, 256, 0, stream>>>(Xp, meanb, varb, M, N);
    long long tot = (long long)M * N;
    bn_act_kernel<<<(unsigned)((tot + 255) / 256), 256, 0, stream>>>(
        Xp, Yp, meanb, varb, gp, betap, resp, M, N, relu);
  };

  // ---- Embedding: loc = gather(points, idx0); feats = relu(bn(loc @ eW + eb))
  {
    int total = B * 1024 * 3;
    gather_kernel<<<(total + 255) / 256, 256, 0, stream>>>(points, idx0, locA, B, 2048, 1024, 3);
    gemm(locA, eW, eb, featB, B * 1024, 3, 64);
    bn(featB, featA, eg, ebeta, nullptr, B * 1024, 64, 1);
  }

  float* locCur = locA;  float* locNxt = locB;
  float* featCur = featA; float* featNxt = featB;

  for (int i = 0; i < 4; ++i) {
    int Np = counts[i], S = counts[i + 1];
    int cin = chans[i], cout = chans[i + 1];
    const int* idx = idxs[i];
    int base = 20 + 30 * i;   // stage keys sorted: alpha, beta, pos, pre, transfer
    const float* alpha = (const float*)d_in[base + 0];
    const float* abeta = (const float*)d_in[base + 1];
    const float* posW[2]    = {(const float*)d_in[base + 2],  (const float*)d_in[base + 6]};
    const float* posb[2]    = {(const float*)d_in[base + 3],  (const float*)d_in[base + 7]};
    const float* posbeta[2] = {(const float*)d_in[base + 4],  (const float*)d_in[base + 8]};
    const float* posg[2]    = {(const float*)d_in[base + 5],  (const float*)d_in[base + 9]};
    const float* c1W[2]     = {(const float*)d_in[base + 10], (const float*)d_in[base + 18]};
    const float* c1b[2]     = {(const float*)d_in[base + 11], (const float*)d_in[base + 19]};
    const float* c1beta[2]  = {(const float*)d_in[base + 12], (const float*)d_in[base + 20]};
    const float* c1g[2]     = {(const float*)d_in[base + 13], (const float*)d_in[base + 21]};
    const float* c2W[2]     = {(const float*)d_in[base + 14], (const float*)d_in[base + 22]};
    const float* c2b[2]     = {(const float*)d_in[base + 15], (const float*)d_in[base + 23]};
    const float* c2beta[2]  = {(const float*)d_in[base + 16], (const float*)d_in[base + 24]};
    const float* c2g[2]     = {(const float*)d_in[base + 17], (const float*)d_in[base + 25]};
    const float* tW    = (const float*)d_in[base + 26];
    const float* tb    = (const float*)d_in[base + 27];
    const float* tbeta = (const float*)d_in[base + 28];
    const float* tg    = (const float*)d_in[base + 29];

    { int total = B * S * 3;
      gather_kernel<<<(total + 255) / 256, 256, 0, stream>>>(locCur, idx, locNxt, B, Np, S, 3); }
    { dim3 g(S, B);
      knn_kernel<<<g, 256, 0, stream>>>(locCur, idx, knn, B, Np, S); }

    zero_kernel<<<1, 16, 0, stream>>>(ssq, 16);
    int XC = 2 * cin + 3;
    { int total = B * S * (cin + 3);
      group_center_kernel<<<(total + 255) / 256, 256, 0, stream>>>(
          featCur, locCur, knn, X, ssq, B, Np, cin, S, XC); }
    { float denom = (float)((long long)S * KNN * (cin + 3) - 1);
      stdinv_kernel<<<1, 16, 0, stream>>>(ssq, stdi, denom, B); }
    { long long total = (long long)B * S * KNN * XC;
      finalize_x_kernel<<<(unsigned)((total + 255) / 256), 256, 0, stream>>>(
          X, alpha, abeta, stdi, featCur, idx, B, S, Np, cin, XC); }

    int M = B * S * KNN;
    gemm(X, tW, tb, Y2, M, XC, cout);
    bn(Y2, Y1, tg, tbeta, nullptr, M, cout, 1);
    for (int j = 0; j < 2; ++j) {
      gemm(Y1, c1W[j], c1b[j], Y2, M, cout, cout);
      bn(Y2, Y2, c1g[j], c1beta[j], nullptr, M, cout, 1);
      gemm(Y2, c2W[j], c2b[j], X, M, cout, cout);      // X reused as scratch
      bn(X, Y1, c2g[j], c2beta[j], Y1, M, cout, 0);    // + residual
    }
    { int total = B * S * cout;
      maxpool_kernel<<<(total + 255) / 256, 256, 0, stream>>>(Y1, poolA, B * S, cout); }
    gemm(poolA, posW[0], posb[0], poolB, B * S, cout, cout);
    bn(poolB, poolB, posg[0], posbeta[0], nullptr, B * S, cout, 1);
    gemm(poolB, posW[1], posb[1], featNxt, B * S, cout, cout);
    bn(featNxt, featNxt, posg[1], posbeta[1], nullptr, B * S, cout, 1);

    float* t;
    t = locCur; locCur = locNxt; locNxt = t;
    t = featCur; featCur = featNxt; featNxt = t;
  }

  // ---- Classifier head
  { int total = B * 1024;
    gmax_kernel<<<(total + 255) / 256, 256, 0, stream>>>(featCur, poolA, B, 64, 1024); }
  gemm(poolA, fc1W, fc1b, poolB, B, 1024, 512);
  bn(poolB, poolB, bn1_g, bn1_beta, nullptr, B, 512, 1);
  gemm(poolB, fc2W, fc2b, poolA, B, 512, 256);
  bn(poolA, poolA, bn2_g, bn2_beta, nullptr, B, 256, 1);
  gemm(poolA, fc3W, fc3b, (float*)d_out, B, 256, 40);
}
